// ModulatedConv2d_22239340658911
// MI455X (gfx1250) — matrix operand order repaired and checked by reading, not compiled, not run on hardware
//
#include <hip/hip_runtime.h>

typedef unsigned int   u32;
typedef unsigned short u16;
typedef __attribute__((ext_vector_type(4)))  u32   v4u;
typedef __attribute__((ext_vector_type(8)))  u32   v8u;
typedef __attribute__((ext_vector_type(8)))  float v8f;
typedef __attribute__((ext_vector_type(16))) __bf16 v16bf;

#define BATCH 16
#define IN_C  512
#define OUT_C 512
#define HW    64
#define BLK_OC 128
#define BLK_ROWS 4
#define ICB    32

// 1/sqrt(512*9), 1/(512*9)
#define WSCALE  1.4731391e-2f
#define WSCALE2 2.17013888889e-4f

// ic-line pitch in dwords: 16 data dwords (32 bf16) + 4 pad -> 80B stride,
// so 16-lane b128 frag loads start at 20n mod 64 (all = 0 mod 4, disjoint,
// covering all 64 banks): conflict-free two-phase b128.
#define XPITCH 20
#define WPITCH 20
#define XROWS  (BLK_ROWS + 2)   // 6 halo row planes

__device__ __forceinline__ void bf16split(float v, u32& hb16, u32& lb16) {
    u32 uv = __float_as_uint(v);
    u32 hb = uv & 0xffff0000u;
    float lf = v - __uint_as_float(hb);   // exact in f32
    hb16 = hb >> 16;
    lb16 = __float_as_uint(lf) >> 16;     // truncate lo to bf16
}

// ---------------- phase 1: style modulation s[b,ic] ----------------
__global__ __launch_bounds__(256) void style_mod_kernel(
    const float* __restrict__ style, const float* __restrict__ mod_w,
    const float* __restrict__ mod_b, float* __restrict__ s)
{
    const int b = blockIdx.x;
    for (int ic = threadIdx.x; ic < IN_C; ic += 256) {
        float acc = mod_b[ic];
        const float* st = style + b * IN_C;
        const float* mw = mod_w + ic * IN_C;
        for (int j = 0; j < IN_C; ++j) acc += st[j] * mw[j];
        s[b * IN_C + ic] = acc;
    }
}

// ---------------- phase 2: wsq[oc,ic] = sum_tap w^2 ----------------
__global__ __launch_bounds__(256) void wsq_kernel(
    const float* __restrict__ weight, float* __restrict__ wsq)
{
    const int idx = blockIdx.x * 256 + threadIdx.x; // oc*512 + ic
    const float* wp = weight + idx * 9;
    float a = 0.0f;
    #pragma unroll
    for (int t = 0; t < 9; ++t) a += wp[t] * wp[t];
    wsq[idx] = a;
}

// ---------------- phase 3: demod d[b,oc] (exact f32) ----------------
__global__ __launch_bounds__(256) void demod_kernel(
    const float* __restrict__ s, const float* __restrict__ wsq,
    float* __restrict__ d)
{
    const int idx = blockIdx.x * 256 + threadIdx.x; // b*512 + oc
    const int b = idx >> 9;
    const int oc = idx & 511;
    const float* sp = s + b * IN_C;
    const float* wp = wsq + oc * IN_C;
    float acc = 0.0f;
    for (int ic = 0; ic < IN_C; ++ic) {
        float sv = sp[ic];
        acc += sv * sv * wp[ic];
    }
    d[idx] = rsqrtf(acc * WSCALE2 + 1e-8f);
}

// ---- phase 4: weight -> bf16 hi/lo planes, layout [tap][oc][ic] ----
__global__ __launch_bounds__(256) void wprep_kernel(
    const float* __restrict__ weight,
    u16* __restrict__ whi, u16* __restrict__ wlo)
{
    const int idx = blockIdx.x * 256 + threadIdx.x;  // [t][oc][ic]
    const int t   = idx >> 18;
    const int rem = idx & 0x3ffff;
    const int oc  = rem >> 9;
    const int ic  = rem & 511;
    const float v = weight[((oc << 9) + ic) * 9 + t] * WSCALE;
    u32 hb, lb;
    bf16split(v, hb, lb);
    whi[idx] = (u16)hb;
    wlo[idx] = (u16)lb;
}

// ---------------- phase 5: bf16x3 WMMA modulated conv ----------------
// block: 128 oc x 256 px (4 rows), 8 waves (2 M x 4 N);
// wave: 64 oc x 64 px (one row) = 4x4 16x16 f32 tiles, 3 bf16 WMMA/tile/tap.
__global__ __launch_bounds__(256) void modconv_wmma_kernel(
    const float* __restrict__ x,
    const u16* __restrict__ whi_g, const u16* __restrict__ wlo_g,
    const float* __restrict__ noise, const float* __restrict__ noise_scale,
    const float* __restrict__ s, const float* __restrict__ dmod,
    float* __restrict__ out)
{
    __shared__ __align__(16) u32 XhiL[XROWS][68][XPITCH];  // 32640 B
    __shared__ __align__(16) u32 XloL[XROWS][68][XPITCH];  // 32640 B
    __shared__ __align__(16) u32 WhiL[BLK_OC][WPITCH];     // 10240 B
    __shared__ __align__(16) u32 WloL[BLK_OC][WPITCH];     // 10240 B

    const int tid = threadIdx.x;
    const int b   = blockIdx.z;
    const int oc0 = blockIdx.y * BLK_OC;
    const int h0  = blockIdx.x * BLK_ROWS;

    const int lane = tid & 31;
    const int wid  = tid >> 5;
    const int wm   = wid >> 2;   // 2 M-groups of 64 oc
    const int wn   = wid & 3;    // 4 N-groups: output row h0+wn
    const int l16  = lane & 15;
    const int hi   = lane >> 4;  // half-wave: upper K half of fragments

    v8f acc[4][4] = {};

    // zero X once: halo cols (0,65), pad dwords and OOB row planes stay 0.
    {
        u32* z1 = &XhiL[0][0][0];
        u32* z2 = &XloL[0][0][0];
        for (int f = tid; f < XROWS * 68 * XPITCH; f += 256) { z1[f] = 0; z2[f] = 0; }
    }

    // W stage: per (chunk, tap) each thread copies 32B/plane via registers
    const int wo    = tid >> 1;
    const int whalf = tid & 1;
    v4u wrh0, wrh1, wrl0, wrl1;
    auto loadW = [&](int ic0, int t) {
        const size_t g = ((size_t)(t * OUT_C + oc0 + wo) * IN_C) + ic0 + whalf * 16;
        wrh0 = *(const v4u*)(whi_g + g);
        wrh1 = *(const v4u*)(whi_g + g + 8);
        wrl0 = *(const v4u*)(wlo_g + g);
        wrl1 = *(const v4u*)(wlo_g + g + 8);
    };
    auto storeW = [&]() {
        *(v4u*)&WhiL[wo][whalf * 8]     = wrh0;
        *(v4u*)&WhiL[wo][whalf * 8 + 4] = wrh1;
        *(v4u*)&WloL[wo][whalf * 8]     = wrl0;
        *(v4u*)&WloL[wo][whalf * 8 + 4] = wrl1;
    };

    // fragment loads (ISA 7.12.2 16-bit layouts, K-pair packed per dword)
    auto loadFragA = [&](const u32* rowbase) -> v16bf {
        v4u d0 = *(const v4u*)(rowbase + hi * 4);       // K0-7  / K8-15
        v4u d1 = *(const v4u*)(rowbase + 8 + hi * 4);   // K16-23/ K24-31
        v8u c = {d0.x, d0.y, d0.z, d0.w, d1.x, d1.y, d1.z, d1.w};
        return __builtin_bit_cast(v16bf, c);
    };
    auto loadFragB = [&](const u32* cellbase) -> v16bf {
        v4u d0 = *(const v4u*)(cellbase + hi * 4);      // K0-7  / K8-15
        v4u d1 = *(const v4u*)(cellbase + 8 + hi * 4);  // K16-23/ K24-31
        v8u c = {d0.x, d0.y, d0.z, d0.w, d1.x, d1.y, d1.z, d1.w};
        return __builtin_bit_cast(v16bf, c);
    };

    for (int ic0 = 0; ic0 < IN_C; ic0 += ICB) {
        __syncthreads();   // previous chunk's readers of X and W are done

        // ---- stage X*s as packed bf16 hi/lo, transposed to ic-major ----
        // 6 row planes x 64 cols = 384 cells over 256 threads (2 passes)
        #pragma unroll
        for (int it = 0; it < 2; ++it) {
            const int cell = tid + it * 256;
            if (cell < XROWS * HW) {
                const int rr   = cell >> 6;
                const int colt = cell & 63;
                const int rowg = h0 - 1 + rr;
                if (rowg >= 0 && rowg < HW) {
                    const size_t gbase =
                        ((size_t)(b * IN_C + ic0) * HW + rowg) * HW + colt;
                    #pragma unroll 4
                    for (int k = 0; k < 16; ++k) {
                        const float sv0 = s[b * IN_C + ic0 + 2 * k];
                        const float sv1 = s[b * IN_C + ic0 + 2 * k + 1];
                        const float v0 = x[gbase + (size_t)(2 * k) * (HW * HW)] * sv0;
                        const float v1 = x[gbase + (size_t)(2 * k + 1) * (HW * HW)] * sv1;
                        u32 hb0, lb0, hb1, lb1;
                        bf16split(v0, hb0, lb0);
                        bf16split(v1, hb1, lb1);
                        XhiL[rr][colt + 1][k] = hb0 | (hb1 << 16);
                        XloL[rr][colt + 1][k] = lb0 | (lb1 << 16);
                    }
                }
            }
        }
        loadW(ic0, 0);

        #pragma unroll
        for (int t = 0; t < 9; ++t) {
            if (t) __syncthreads();        // readers of previous tap's W done
            storeW();
            if (t < 8) loadW(ic0, t + 1);  // prefetch next tap under compute
            __syncthreads();               // W (and X at t=0) visible

            const int kh = t / 3;
            const int kw = t - 3 * kh;
            const int r  = wn + kh;        // stored row plane
            const int c0 = l16 + kw;       // shifted column for this tap

            v16bf Ah[4], Al[4];
            #pragma unroll
            for (int mt = 0; mt < 4; ++mt) {
                const int orow = wm * 64 + mt * 16 + l16;
                Ah[mt] = loadFragA(&WhiL[orow][0]);
                Al[mt] = loadFragA(&WloL[orow][0]);
            }
            #pragma unroll
            for (int nt = 0; nt < 4; ++nt) {
                const int c = c0 + nt * 16;
                const v16bf Bh = loadFragB(&XhiL[r][c][0]);
                const v16bf Bl = loadFragB(&XloL[r][c][0]);
                #pragma unroll
                for (int mt = 0; mt < 4; ++mt) {
                    acc[mt][nt] = __builtin_amdgcn_wmma_f32_16x16x32_bf16(false, Ah[mt], false, Bl, (short)0, acc[mt][nt], false, false);
                    acc[mt][nt] = __builtin_amdgcn_wmma_f32_16x16x32_bf16(false, Al[mt], false, Bh, (short)0, acc[mt][nt], false, false);
                    acc[mt][nt] = __builtin_amdgcn_wmma_f32_16x16x32_bf16(false, Ah[mt], false, Bh, (short)0, acc[mt][nt], false, false);
                }
            }
        }
    }

    // ---- epilogue: * d[b,oc] + noise*ns, leaky relu(0.2) ----
    const float nsv  = noise_scale[0];
    const int   hrow = h0 + wn;
    #pragma unroll
    for (int nt = 0; nt < 4; ++nt) {
        const int col = nt * 16 + l16;
        const float nz = noise[(b * HW + hrow) * HW + col] * nsv;
        #pragma unroll
        for (int mt = 0; mt < 4; ++mt) {
            #pragma unroll
            for (int v = 0; v < 8; ++v) {
                const int oc = oc0 + wm * 64 + mt * 16 + hi * 8 + v;
                float val = acc[mt][nt][v] * dmod[b * OUT_C + oc] + nz;
                out[((size_t)(b * OUT_C + oc) * HW + hrow) * HW + col] =
                    (val >= 0.0f) ? val : 0.2f * val;
            }
        }
    }
}

extern "C" void kernel_launch(void* const* d_in, const int* in_sizes, int n_in,
                              void* d_out, int out_size, void* d_ws, size_t ws_size,
                              hipStream_t stream) {
    const float* x           = (const float*)d_in[0];
    const float* style       = (const float*)d_in[1];
    const float* noise       = (const float*)d_in[2];
    const float* weight      = (const float*)d_in[3];
    const float* mod_w       = (const float*)d_in[4];
    const float* mod_b       = (const float*)d_in[5];
    const float* noise_scale = (const float*)d_in[6];
    float* out = (float*)d_out;

    float* ws      = (float*)d_ws;
    float* s_buf   = ws;                       // 16*512 f32
    float* d_buf   = ws + BATCH * IN_C;        // 16*512 f32
    float* wsq_buf = ws + 2 * BATCH * IN_C;    // 512*512 f32
    u16*   whi     = (u16*)(ws + 2 * BATCH * IN_C + OUT_C * IN_C); // 16B-aligned
    u16*   wlo     = whi + 9 * OUT_C * IN_C;
    // total workspace: ~10.55 MB

    style_mod_kernel<<<BATCH, 256, 0, stream>>>(style, mod_w, mod_b, s_buf);
    wsq_kernel<<<(OUT_C * IN_C) / 256, 256, 0, stream>>>(weight, wsq_buf);
    demod_kernel<<<(BATCH * OUT_C) / 256, 256, 0, stream>>>(s_buf, wsq_buf, d_buf);
    wprep_kernel<<<(9 * OUT_C * IN_C) / 256, 256, 0, stream>>>(weight, whi, wlo);

    dim3 grid(HW / BLK_ROWS, OUT_C / BLK_OC, BATCH);
    modconv_wmma_kernel<<<grid, 256, 0, stream>>>(x, whi, wlo, noise, noise_scale,
                                                  s_buf, d_buf, out);
}
